// GAT_42597485642263
// MI455X (gfx1250) — matrix operand order repaired and checked by reading, not compiled, not run on hardware
//
#include <hip/hip_runtime.h>
#include <math.h>

typedef __attribute__((ext_vector_type(16))) _Float16 v16h;
typedef __attribute__((ext_vector_type(8)))  float    v8f;

#define NEG_SLOPE 0.2f

// ---------------------------------------------------------------- utilities

__global__ void fill_f32(float* __restrict__ p, float v, long long n) {
    long long i = (long long)blockIdx.x * blockDim.x + threadIdx.x;
    if (i < n) p[i] = v;
}

__device__ __forceinline__ void atomicMaxFloat(float* addr, float val) {
    // sign-magnitude ordering trick; buffer initialized to -inf
    if (val >= 0.0f)
        atomicMax((int*)addr, __float_as_int(val));
    else
        atomicMin((unsigned int*)addr, (unsigned int)__float_as_int(val));
}

// ---------------------------------------------------------------- WMMA GEMM
// C[n, TN*16] = A[n,128] @ B[128, TN*16], row-major f32.
// Block = 256 threads (8 waves). B panel is staged once per block into LDS,
// converted to f16 and pre-swizzled into V_WMMA B-fragment order, so each
// fragment read is one contiguous 32B ds_load per lane. Each wave computes a
// 16 x (TN*16) output strip with TN independent accumulators -> TN
// back-to-back v_wmma_f32_16x16x32_f16 per K-step (no RAW hazard stalls).
template <int TN>
__global__ void __launch_bounds__(256)
gemm_wmma_lds(const float* __restrict__ A, const float* __restrict__ B,
              float* __restrict__ C, int n) {
    constexpr int K  = 128;
    constexpr int M  = TN * 16;
    constexpr int KB = K / 32;                       // 4 K-steps
    __shared__ _Float16 Bsh[TN * KB * 32 * 16];      // fragment-ordered panel

    // ---- stage B: f32 row-major [K][M] -> f16 fragment order ----
    // Fragment (ct, kb): lane holds rows kb*32 + (lane<16 ? 0 : 16) + j,
    // column ct*16 + (lane&15), j = 0..15, packed contiguously (32B/lane).
    for (int idx = threadIdx.x; idx < K * M; idx += 256) {
        int r  = idx / M, c = idx - r * M;           // coalesced in c
        int ct = c >> 4;
        int kb = r >> 5;
        int kr = r & 31;
        int ln = (c & 15) + ((kr & 16) ? 16 : 0);
        int j  = kr & 15;
        Bsh[(((ct * KB + kb) * 32) + ln) * 16 + j] = (_Float16)B[idx];
    }
    __syncthreads();

    int lane = threadIdx.x & 31;
    int rt   = blockIdx.x * 8 + (threadIdx.x >> 5); // row tile, wave-uniform
    if (rt * 16 >= n) return;                        // after barrier: safe
    int laneLo = lane & 15;
    int hi     = lane >> 4;

    v8f acc[TN];
    const v8f zero = {0.f, 0.f, 0.f, 0.f, 0.f, 0.f, 0.f, 0.f};
#pragma unroll
    for (int t = 0; t < TN; ++t) acc[t] = zero;

    const float* arow = A + (long long)(rt * 16 + laneLo) * K;
#pragma unroll
    for (int kb = 0; kb < KB; ++kb) {
        // A fragment (16-bit, 16x32): lane = row; halves 0-15 = K[ka,ka+16);
        // ka = kb*32 + 16*hi. Coalesced b128 loads.
        int ka = kb * 32 + (hi ? 16 : 0);
        float4 a0 = *(const float4*)(arow + ka);
        float4 a1 = *(const float4*)(arow + ka + 4);
        float4 a2 = *(const float4*)(arow + ka + 8);
        float4 a3 = *(const float4*)(arow + ka + 12);
        v16h a;
        a[0]  = (_Float16)a0.x; a[1]  = (_Float16)a0.y;
        a[2]  = (_Float16)a0.z; a[3]  = (_Float16)a0.w;
        a[4]  = (_Float16)a1.x; a[5]  = (_Float16)a1.y;
        a[6]  = (_Float16)a1.z; a[7]  = (_Float16)a1.w;
        a[8]  = (_Float16)a2.x; a[9]  = (_Float16)a2.y;
        a[10] = (_Float16)a2.z; a[11] = (_Float16)a2.w;
        a[12] = (_Float16)a3.x; a[13] = (_Float16)a3.y;
        a[14] = (_Float16)a3.z; a[15] = (_Float16)a3.w;
#pragma unroll
        for (int ct = 0; ct < TN; ++ct) {
            v16h b = *(const v16h*)&Bsh[(((ct * KB + kb) * 32) + lane) * 16];
            acc[ct] = __builtin_amdgcn_wmma_f32_16x16x32_f16(
                false, a, false, b, (short)0, acc[ct], false, false);
        }
    }

    // C/D f32 layout: VGPR r -> row rt*16 + r + 8*hi, col ct*16 + laneLo
    float* cbase = C + (long long)(rt * 16 + (hi ? 8 : 0)) * M + laneLo;
#pragma unroll
    for (int ct = 0; ct < TN; ++ct) {
        float* crow = cbase + ct * 16;
#pragma unroll
        for (int r = 0; r < 8; ++r) crow[(long long)r * M] = acc[ct][r];
    }
}

// ---------------------------------------------------------------- GAT pieces

// per-node attention scores: hs[i] = h[i,:].a_s ; hd[i] = h[i,:].a_d  (wave per node)
__global__ void node_scores(const float* __restrict__ h,
                            const float* __restrict__ as_, const float* __restrict__ ad_,
                            float* __restrict__ hs, float* __restrict__ hd, int n) {
    int wave = (int)((blockIdx.x * (long long)blockDim.x + threadIdx.x) >> 5);
    int lane = threadIdx.x & 31;
    if (wave >= n) return;
    const float* row = h + (long long)wave * 128;
    float s = 0.f, d = 0.f;
#pragma unroll
    for (int c = 0; c < 128; c += 32) {
        float v = row[c + lane];
        s += v * as_[c + lane];
        d += v * ad_[c + lane];
    }
#pragma unroll
    for (int off = 16; off; off >>= 1) {
        s += __shfl_down(s, off);
        d += __shfl_down(d, off);
    }
    if (lane == 0) { hs[wave] = s; hd[wave] = d; }
}

__device__ __forceinline__ void edge_endpoints(const int* __restrict__ ei, int E, int id,
                                               int& src, int& dst) {
    if (id < E) { src = ei[id]; dst = ei[E + id]; }
    else        { src = dst = id - E; }          // appended self-loops
}

__device__ __forceinline__ float lrelu(float v) {
    return v > 0.f ? v : NEG_SLOPE * v;
}

// pass 1: segment max of leaky_relu score over dst  (thread per edge)
__global__ void edge_max_k(const int* __restrict__ ei, int E, int n,
                           const float* __restrict__ hs, const float* __restrict__ hd,
                           float* __restrict__ mx) {
    int id = blockIdx.x * blockDim.x + threadIdx.x;
    if (id >= E + n) return;
    int src, dst; edge_endpoints(ei, E, id, src, dst);
    atomicMaxFloat(&mx[dst], lrelu(hs[src] + hd[dst]));
}

// pass 2: segment sum of exp(e - max)  (thread per edge; score recomputed)
__global__ void edge_sum_k(const int* __restrict__ ei, int E, int n,
                           const float* __restrict__ hs, const float* __restrict__ hd,
                           const float* __restrict__ mx, float* __restrict__ sm) {
    int id = blockIdx.x * blockDim.x + threadIdx.x;
    if (id >= E + n) return;
    int src, dst; edge_endpoints(ei, E, id, src, dst);
    float p = expf(lrelu(hs[src] + hd[dst]) - mx[dst]);
    atomicAdd(&sm[dst], p);
}

// pass 3: agg[dst,:] += alpha * h[src,:]  (wave per edge, float4 per lane)
__global__ void edge_agg_k(const int* __restrict__ ei, int E, int n,
                           const float* __restrict__ hs, const float* __restrict__ hd,
                           const float* __restrict__ mx, const float* __restrict__ sm,
                           const float* __restrict__ h, float* __restrict__ agg) {
    int wave = (int)((blockIdx.x * (long long)blockDim.x + threadIdx.x) >> 5);
    int lane = threadIdx.x & 31;
    if (wave >= E + n) return;
    int src, dst; edge_endpoints(ei, E, wave, src, dst);
    float alpha = expf(lrelu(hs[src] + hd[dst]) - mx[dst]) / sm[dst];
    float4 v = ((const float4*)(h + (long long)src * 128))[lane];
    float* a = agg + (long long)dst * 128 + lane * 4;
    atomicAdd(a + 0, v.x * alpha);
    atomicAdd(a + 1, v.y * alpha);
    atomicAdd(a + 2, v.z * alpha);
    atomicAdd(a + 3, v.w * alpha);
}

__global__ void bias_relu_k(float* __restrict__ io, const float* __restrict__ b,
                            long long total, int c) {
    long long i = (long long)blockIdx.x * blockDim.x + threadIdx.x;
    if (i >= total) return;
    float v = io[i] + b[i & (c - 1)];          // c is a power of two (128/64)
    io[i] = v > 0.f ? v : 0.f;
}

__global__ void bias_add_k(float* __restrict__ io, const float* __restrict__ b,
                           long long total, int c) {
    long long i = (long long)blockIdx.x * blockDim.x + threadIdx.x;
    if (i >= total) return;
    io[i] = io[i] + b[i & (c - 1)];
}

// ---------------------------------------------------------------- GCN pieces

__global__ void deg_count_k(const int* __restrict__ ei, int E, float* __restrict__ deg) {
    int id = blockIdx.x * blockDim.x + threadIdx.x;
    if (id >= E) return;
    atomicAdd(&deg[ei[E + id]], 1.0f);
}

__global__ void rsqrt_k(float* __restrict__ d, int n) {
    int i = blockIdx.x * blockDim.x + threadIdx.x;
    if (i < n) d[i] = rsqrtf(d[i]);
}

// zagg[dst,:] += dinv[src]*dinv[dst] * hw[src,:]  (wave per edge, float2 per lane, 64 ch)
__global__ void gcn_agg_k(const int* __restrict__ ei, int E, int n,
                          const float* __restrict__ dinv, const float* __restrict__ hw,
                          float* __restrict__ zagg) {
    int wave = (int)((blockIdx.x * (long long)blockDim.x + threadIdx.x) >> 5);
    int lane = threadIdx.x & 31;
    if (wave >= E + n) return;
    int src, dst; edge_endpoints(ei, E, wave, src, dst);
    float norm = dinv[src] * dinv[dst];
    float2 v = ((const float2*)(hw + (long long)src * 64))[lane];
    float* a = zagg + (long long)dst * 64 + lane * 2;
    atomicAdd(a + 0, v.x * norm);
    atomicAdd(a + 1, v.y * norm);
}

// out[l] = dot(z[eli0[l],:], z[eli1[l],:]) over 64 channels  (wave per pair)
__global__ void decode_k(const int* __restrict__ eli, int nl,
                         const float* __restrict__ z, float* __restrict__ out) {
    int wave = (int)((blockIdx.x * (long long)blockDim.x + threadIdx.x) >> 5);
    int lane = threadIdx.x & 31;
    if (wave >= nl) return;
    int a = eli[wave], b = eli[nl + wave];
    float2 za = ((const float2*)(z + (long long)a * 64))[lane];
    float2 zb = ((const float2*)(z + (long long)b * 64))[lane];
    float s = za.x * zb.x + za.y * zb.y;
#pragma unroll
    for (int off = 16; off; off >>= 1) s += __shfl_down(s, off);
    if (lane == 0) out[wave] = s;
}

// ---------------------------------------------------------------- driver

static inline unsigned blocks_for(long long threads) {
    return (unsigned)((threads + 255) / 256);
}

extern "C" void kernel_launch(void* const* d_in, const int* in_sizes, int n_in,
                              void* d_out, int out_size, void* d_ws, size_t ws_size,
                              hipStream_t stream) {
    const float* x   = (const float*)d_in[0];
    const int*   ei  = (const int*)  d_in[1];
    const int*   eli = (const int*)  d_in[2];
    const float* W[3]  = { (const float*)d_in[3], (const float*)d_in[7],  (const float*)d_in[11] };
    const float* As[3] = { (const float*)d_in[4], (const float*)d_in[8],  (const float*)d_in[12] };
    const float* Ad[3] = { (const float*)d_in[5], (const float*)d_in[9],  (const float*)d_in[13] };
    const float* Bi[3] = { (const float*)d_in[6], (const float*)d_in[10], (const float*)d_in[14] };
    const float* W4 = (const float*)d_in[15];
    const float* b4 = (const float*)d_in[16];
    float* out = (float*)d_out;

    const int N  = in_sizes[0] / 128;
    const int E  = in_sizes[1] / 2;
    const int NL = in_sizes[2] / 2;
    const int Ep = E + N;                       // edges incl. self-loops
    const long long NB = (long long)N * 128;

    // workspace layout (floats): 3 rotating N x 128 buffers + per-node scalars
    float* ws = (float*)d_ws;
    float* B0 = ws;
    float* B1 = ws + NB;
    float* B2 = ws + 2 * NB;
    float* hs  = ws + 3 * NB;
    float* hd  = hs + N;
    float* mx  = hd + N;
    float* sm  = mx + N;
    float* deg = sm + N;

    const dim3 blk(256);
    const float NEG_INF = -__builtin_huge_valf();
    const unsigned gemmBlocks = (unsigned)((N / 16 + 7) / 8);  // 8 row tiles / block

    // buffer rotation: in -> (h, agg); agg becomes next layer's input after bias+relu
    const float* cur = x;
    float* rot[3] = { B0, B1, B2 };
    int hI = 0, gI = 1, fI = 2;

    for (int l = 0; l < 3; ++l) {
        float* h   = rot[hI];
        float* agg = rot[gI];

        fill_f32<<<blocks_for(NB), blk, 0, stream>>>(agg, 0.0f, NB);
        fill_f32<<<blocks_for(N),  blk, 0, stream>>>(mx, NEG_INF, N);
        fill_f32<<<blocks_for(N),  blk, 0, stream>>>(sm, 0.0f, N);

        gemm_wmma_lds<8><<<gemmBlocks, blk, 0, stream>>>(cur, W[l], h, N);
        node_scores<<<blocks_for((long long)N * 32), blk, 0, stream>>>(h, As[l], Ad[l], hs, hd, N);
        edge_max_k<<<blocks_for(Ep), blk, 0, stream>>>(ei, E, N, hs, hd, mx);
        edge_sum_k<<<blocks_for(Ep), blk, 0, stream>>>(ei, E, N, hs, hd, mx, sm);
        edge_agg_k<<<blocks_for((long long)Ep * 32), blk, 0, stream>>>(ei, E, N, hs, hd, mx, sm, h, agg);
        bias_relu_k<<<blocks_for(NB), blk, 0, stream>>>(agg, Bi[l], NB, 128);

        cur = agg;
        int newF = hI;        // h buffer is dead -> becomes free
        hI = fI;              // old free becomes next h
        fI = newF;
        gI = (0 + 1 + 2) - hI - fI;
    }

    // GCN layer: hw = cur @ W4 (N x 64); zagg = normalized scatter; z = zagg + b4
    float* hw   = rot[hI];
    float* zagg = rot[fI];
    const long long NZ = (long long)N * 64;

    fill_f32<<<blocks_for(NZ), blk, 0, stream>>>(zagg, 0.0f, NZ);
    fill_f32<<<blocks_for(N),  blk, 0, stream>>>(deg, 1.0f, N);   // self-loop
    deg_count_k<<<blocks_for(E), blk, 0, stream>>>(ei, E, deg);
    rsqrt_k<<<blocks_for(N), blk, 0, stream>>>(deg, N);           // deg -> dinv in place

    gemm_wmma_lds<4><<<gemmBlocks, blk, 0, stream>>>(cur, W4, hw, N);
    gcn_agg_k<<<blocks_for((long long)Ep * 32), blk, 0, stream>>>(ei, E, N, deg, hw, zagg);
    bias_add_k<<<blocks_for(NZ), blk, 0, stream>>>(zagg, b4, NZ, 64);

    // link decode
    decode_k<<<blocks_for((long long)NL * 32), blk, 0, stream>>>(eli, NL, zagg, out);
}
